// NodeAggregator_38809324486702
// MI455X (gfx1250) — matrix operand order repaired and checked by reading, not compiled, not run on hardware
//
#include <hip/hip_runtime.h>

// ---------------------------------------------------------------------------
// NodeAggregator (DiffPool-style) for MI455X / gfx1250, wave32 + WMMA bf16.
//
// Roofline: adj stream = 268 MB @ 23.3 TB/s ~= 11.5us lower bound; adj@S has
// 32 flop/byte on that stream -> matrix-core rate required, so adj tiles are
// converted f32->bf16 in-flight (native cvt, not manual RNE bit ops) and fed
// to v_wmma_f32_16x16x32_bf16. S and mid are staged in scratch as bf16,
// column-major, so WMMA A/B fragments are contiguous b128 loads
// (k = 16*(lane>>4) + s, same order in A and B).
// ---------------------------------------------------------------------------

typedef __attribute__((ext_vector_type(16))) __bf16 v16bf;
typedef __attribute__((ext_vector_type(8)))  float  v8f;
typedef __bf16       bf16x4 __attribute__((ext_vector_type(4)));
typedef __bf16       bf16x8 __attribute__((ext_vector_type(8)));
typedef unsigned int u32x4  __attribute__((ext_vector_type(4)));
typedef float        f32x4  __attribute__((ext_vector_type(4)));

union BFrag {
  u32x4  q[2];
  v16bf  v;
  __bf16 s[16];
  bf16x4 h4[4];
};

union OFrag {
  bf16x8 h;
  u32x4  q;
};

#define WMMA_BF16(A, B, C) \
  __builtin_amdgcn_wmma_f32_16x16x32_bf16(false, (A), false, (B), (short)0, (C), false, false)

// A fragment (16x32 bf16, M=lane&15): halves 0..15 = K base+0..15,
// where base already includes (lane>>4)*16. Source is f32;
// conversion via native bf16 cvt (vectorized fptrunc).
__device__ __forceinline__ v16bf loadA_f32(const float* p) {
  BFrag A;
  A.h4[0] = __builtin_convertvector(*(const f32x4*)(p),      bf16x4);
  A.h4[1] = __builtin_convertvector(*(const f32x4*)(p + 4),  bf16x4);
  A.h4[2] = __builtin_convertvector(*(const f32x4*)(p + 8),  bf16x4);
  A.h4[3] = __builtin_convertvector(*(const f32x4*)(p + 12), bf16x4);
  return A.v;
}

// Same A fragment but from pre-converted bf16 (contiguous p[0..15]).
__device__ __forceinline__ v16bf loadA_bf16(const __bf16* p) {
  BFrag A;
  A.q[0] = *(const u32x4*)(p);
  A.q[1] = *(const u32x4*)(p + 8);
  return A.v;
}

// B fragment (32x16 bf16, col=lane&15): 16 contiguous K values per lane.
__device__ __forceinline__ v16bf loadB_bf16(const __bf16* p) {
  BFrag B;
  B.q[0] = *(const u32x4*)(p);
  B.q[1] = *(const u32x4*)(p + 8);
  return B.v;
}

// ---------------------------------------------------------------------------
// Prep: W1 [F=128, K=64] f32 row-major -> W1cs bf16 [c][f] (c-major, f contig)
// ---------------------------------------------------------------------------
__global__ void k_prep_w1(const float* __restrict__ W1, __bf16* __restrict__ W1cs) {
  int i = blockIdx.x * blockDim.x + threadIdx.x;
  if (i < 128 * 64) {
    int c = i >> 7, f = i & 127;
    W1cs[c * 128 + f] = (__bf16)W1[f * 64 + c];
  }
}

// ---------------------------------------------------------------------------
// Stage 1: S = softmax(relu(x@W1+b1)@W2 + b2 + maskconst)
// Block = 256 thr (8 waves). Phase 1: WMMA x@W1 for 128 rows (16/wave).
// Phase 2: scalar 64x64 GEMM + softmax per row, W2 in LDS.
// Output: S_cm bf16 [b][c in 64][n in 2048]   (column-major over clusters)
// ---------------------------------------------------------------------------
__global__ __launch_bounds__(256) void k_assign(
    const float* __restrict__ x, const float* __restrict__ mask,
    const __bf16* __restrict__ W1cs, const float* __restrict__ b1,
    const float* __restrict__ W2, const float* __restrict__ b2,
    __bf16* __restrict__ S_cm) {
  __shared__ float h_lds[128 * 64];
  __shared__ float W2_lds[64 * 64];
  __shared__ float b2_lds[64];

  const int tid  = threadIdx.x;
  const int wave = tid >> 5, lane = tid & 31;
  const int l = lane & 15, hi = lane >> 4;
  const int rBase = blockIdx.x * 128;
  const int r0 = wave * 16;

  for (int i = tid; i < 64 * 64; i += 256) W2_lds[i] = W2[i];
  if (tid < 64) b2_lds[tid] = b2[tid];

  // ---- phase 1: h = relu(x @ W1 + b1), wave computes rows [r0, r0+16) ----
  const float* xr = x + (size_t)(rBase + r0 + l) * 128;
  v8f a0 = {}, a1 = {}, a2 = {}, a3 = {};
  #pragma unroll
  for (int f0 = 0; f0 < 128; f0 += 32) {
    v16bf A = loadA_f32(xr + f0 + hi * 16);
    const __bf16* w = W1cs + f0 + hi * 16;
    a0 = WMMA_BF16(A, loadB_bf16(w + (size_t)(l)      * 128), a0);
    a1 = WMMA_BF16(A, loadB_bf16(w + (size_t)(16 + l) * 128), a1);
    a2 = WMMA_BF16(A, loadB_bf16(w + (size_t)(32 + l) * 128), a2);
    a3 = WMMA_BF16(A, loadB_bf16(w + (size_t)(48 + l) * 128), a3);
  }
  {
    float b0v = b1[l], b1v = b1[16 + l], b2v = b1[32 + l], b3v = b1[48 + l];
    #pragma unroll
    for (int v = 0; v < 8; ++v) {
      int row = r0 + v + 8 * hi;
      float h0 = a0[v] + b0v; h_lds[row * 64 + l]      = h0 > 0.f ? h0 : 0.f;
      float h1 = a1[v] + b1v; h_lds[row * 64 + 16 + l] = h1 > 0.f ? h1 : 0.f;
      float h2 = a2[v] + b2v; h_lds[row * 64 + 32 + l] = h2 > 0.f ? h2 : 0.f;
      float h3 = a3[v] + b3v; h_lds[row * 64 + 48 + l] = h3 > 0.f ? h3 : 0.f;
    }
  }
  __syncthreads();

  // ---- phase 2: logits = h@W2 + b2 (+mask const), softmax over 64 ----
  if (tid < 128) {
    const int R = rBase + tid;
    const int b = R >> 11, n = R & 2047;
    float lg[64];
    #pragma unroll
    for (int j = 0; j < 64; ++j) lg[j] = b2_lds[j];
    for (int c = 0; c < 64; ++c) {
      float hc = h_lds[tid * 64 + c];
      #pragma unroll
      for (int j = 0; j < 64; ++j) lg[j] = fmaf(hc, W2_lds[c * 64 + j], lg[j]);
    }
    float mc = -1e9f * (1.0f - mask[R]);
    float mx = -3.0e38f;
    for (int j = 0; j < 64; ++j) { lg[j] += mc; mx = lg[j] > mx ? lg[j] : mx; }
    float sum = 0.f;
    for (int j = 0; j < 64; ++j) { float e = __expf(lg[j] - mx); lg[j] = e; sum += e; }
    float inv = 1.0f / sum;
    __bf16* Sb = S_cm + (size_t)b * 64 * 2048;
    for (int j = 0; j < 64; ++j) Sb[(size_t)j * 2048 + n] = (__bf16)(lg[j] * inv);
  }
}

// ---------------------------------------------------------------------------
// Hot kernel: mid = adj @ S.   One wave = 16 adj rows x all 64 clusters.
// Per k-step: 64 B/lane of adj (HBM stream) -> 4 WMMAs = 32 flop/byte.
// S B-fragments are contiguous bf16 (L2-resident). Output mid_cm bf16 [b][c][n].
// ---------------------------------------------------------------------------
__global__ __launch_bounds__(256) void k_mid(
    const float* __restrict__ adj, const __bf16* __restrict__ S_cm,
    __bf16* __restrict__ mid_cm) {
  const int tid = threadIdx.x;
  const int s = blockIdx.x * 8 + (tid >> 5);     // strip id 0..2047
  const int b  = s >> 7;
  const int n0 = (s & 127) << 4;
  const int lane = tid & 31, l = lane & 15, hi = lane >> 4;

  const float* ar = adj + ((size_t)b * 2048 + n0 + l) * 2048;
  const __bf16* Sb = S_cm + (size_t)b * 64 * 2048;
  const __bf16* S0 = Sb + (size_t)(l)      * 2048 + hi * 16;
  const __bf16* S1 = Sb + (size_t)(16 + l) * 2048 + hi * 16;
  const __bf16* S2 = Sb + (size_t)(32 + l) * 2048 + hi * 16;
  const __bf16* S3 = Sb + (size_t)(48 + l) * 2048 + hi * 16;

  v8f c0 = {}, c1 = {}, c2 = {}, c3 = {};
  for (int m0 = 0; m0 < 2048; m0 += 32) {
    __builtin_prefetch(ar + m0 + 256, 0, 0);     // global_prefetch_b8 ahead on adj
    v16bf A = loadA_f32(ar + m0 + hi * 16);
    c0 = WMMA_BF16(A, loadB_bf16(S0 + m0), c0);
    c1 = WMMA_BF16(A, loadB_bf16(S1 + m0), c1);
    c2 = WMMA_BF16(A, loadB_bf16(S2 + m0), c2);
    c3 = WMMA_BF16(A, loadB_bf16(S3 + m0), c3);
  }

  // store mid column-major bf16: lane's 8 C values are contiguous in n -> b128
  __bf16* Mb = mid_cm + (size_t)b * 64 * 2048;
  OFrag O;
  O.h = __builtin_convertvector(c0, bf16x8);
  *(u32x4*)(Mb + (size_t)(l)      * 2048 + n0 + 8 * hi) = O.q;
  O.h = __builtin_convertvector(c1, bf16x8);
  *(u32x4*)(Mb + (size_t)(16 + l) * 2048 + n0 + 8 * hi) = O.q;
  O.h = __builtin_convertvector(c2, bf16x8);
  *(u32x4*)(Mb + (size_t)(32 + l) * 2048 + n0 + 8 * hi) = O.q;
  O.h = __builtin_convertvector(c3, bf16x8);
  *(u32x4*)(Mb + (size_t)(48 + l) * 2048 + n0 + 8 * hi) = O.q;
}

// ---------------------------------------------------------------------------
// pooled_x = S^T @ x  -> [B, 64, 128] f32.  One wave per 16x16 output tile.
// ---------------------------------------------------------------------------
__global__ __launch_bounds__(32) void k_pooled_x(
    const __bf16* __restrict__ S_cm, const float* __restrict__ x,
    float* __restrict__ out_px) {
  const int t = blockIdx.x;                      // 0 .. 16*32-1
  const int b = t >> 5;
  const int c1_0 = ((t >> 3) & 3) * 16;
  const int f0 = (t & 7) * 16;
  const int lane = threadIdx.x & 31, l = lane & 15, hi = lane >> 4;

  const __bf16* Sr = S_cm + ((size_t)b * 64 + c1_0 + l) * 2048 + hi * 16;
  const float* xb = x + (size_t)b * 2048 * 128 + f0 + l;

  v8f acc = {};
  for (int m0 = 0; m0 < 2048; m0 += 32) {
    v16bf A = loadA_bf16(Sr + m0);
    BFrag Bf;
    const float* xp = xb + (size_t)(m0 + hi * 16) * 128;
    #pragma unroll
    for (int i = 0; i < 16; ++i) Bf.s[i] = (__bf16)xp[(size_t)i * 128];
    acc = WMMA_BF16(A, Bf.v, acc);
  }
  float* o = out_px + (size_t)b * 64 * 128 + (size_t)(c1_0 + 8 * hi) * 128 + f0 + l;
  #pragma unroll
  for (int v = 0; v < 8; ++v) o[(size_t)v * 128] = acc[v];
}

// ---------------------------------------------------------------------------
// pooled_adj = S^T @ mid -> [B, 64, 64] f32.  One wave per 16x16 tile.
// ---------------------------------------------------------------------------
__global__ __launch_bounds__(32) void k_pooled_adj(
    const __bf16* __restrict__ S_cm, const __bf16* __restrict__ mid_cm,
    float* __restrict__ out_pa) {
  const int t = blockIdx.x;                      // 0 .. 16*16-1
  const int b = t >> 4;
  const int c1_0 = ((t >> 2) & 3) * 16;
  const int c2_0 = (t & 3) * 16;
  const int lane = threadIdx.x & 31, l = lane & 15, hi = lane >> 4;

  const __bf16* Sr = S_cm   + ((size_t)b * 64 + c1_0 + l) * 2048 + hi * 16;
  const __bf16* Mr = mid_cm + ((size_t)b * 64 + c2_0 + l) * 2048 + hi * 16;

  v8f acc = {};
  for (int m0 = 0; m0 < 2048; m0 += 32) {
    acc = WMMA_BF16(loadA_bf16(Sr + m0), loadB_bf16(Mr + m0), acc);
  }
  float* o = out_pa + (size_t)b * 64 * 64 + (size_t)(c1_0 + 8 * hi) * 64 + c2_0 + l;
  #pragma unroll
  for (int v = 0; v < 8; ++v) o[(size_t)v * 64] = acc[v];
}

__global__ void k_pmask(float* __restrict__ out_pm) {
  int i = blockIdx.x * blockDim.x + threadIdx.x;
  if (i < 16 * 64) out_pm[i] = 1.0f;
}

// ---------------------------------------------------------------------------
extern "C" void kernel_launch(void* const* d_in, const int* in_sizes, int n_in,
                              void* d_out, int out_size, void* d_ws, size_t ws_size,
                              hipStream_t stream) {
  const float* x    = (const float*)d_in[0];
  const float* adj  = (const float*)d_in[1];
  const float* mask = (const float*)d_in[2];
  const float* W1   = (const float*)d_in[3];
  const float* b1   = (const float*)d_in[4];
  const float* W2   = (const float*)d_in[5];
  const float* b2   = (const float*)d_in[6];

  float* out    = (float*)d_out;
  float* out_px = out;                               // 16*64*128
  float* out_pa = out + 16 * 64 * 128;               // 16*64*64
  float* out_pm = out + 16 * 64 * 128 + 16 * 64 * 64;

  // workspace: S_cm (4 MB bf16) | mid_cm (4 MB bf16) | W1cs (16 KB bf16)
  __bf16* S_cm   = (__bf16*)d_ws;
  __bf16* mid_cm = S_cm + (size_t)16 * 64 * 2048;
  __bf16* W1cs   = mid_cm + (size_t)16 * 64 * 2048;

  k_prep_w1   <<<dim3(32),  dim3(256), 0, stream>>>(W1, W1cs);
  k_assign    <<<dim3(256), dim3(256), 0, stream>>>(x, mask, W1cs, b1, W2, b2, S_cm);
  k_mid       <<<dim3(256), dim3(256), 0, stream>>>(adj, S_cm, mid_cm);
  k_pooled_x  <<<dim3(512), dim3(32),  0, stream>>>(S_cm, x, out_px);
  k_pooled_adj<<<dim3(256), dim3(32),  0, stream>>>(S_cm, mid_cm, out_pa);
  k_pmask     <<<dim3(4),   dim3(256), 0, stream>>>(out_pm);
}
